// SCAN_attention_23072564314577
// MI455X (gfx1250) — matrix operand order repaired and checked by reading, not compiled, not run on hardware
//
#include <hip/hip_runtime.h>
#include <math.h>

// ---------------------------------------------------------------------------
// SCAN attention for MI455X (gfx1250), bf16 WMMA pipeline, fp32 accumulate.
// B=128, QL=512, SL=1024, D=1024 (hardcoded from the reference).
// Proj/score GEMMs: B-operand staged through double-buffered LDS with
// global_load_async_to_lds_b128 (ASYNCcnt) overlapped under WMMAs.
// ---------------------------------------------------------------------------

typedef __bf16 bf16_t;
typedef bf16_t v16bf __attribute__((ext_vector_type(16)));
typedef bf16_t v8bf  __attribute__((ext_vector_type(8)));
typedef float  v8f   __attribute__((ext_vector_type(8)));
typedef float  v4f   __attribute__((ext_vector_type(4)));

#define Bx   128
#define QL   512
#define SL   1024
#define DD   1024
#define EPSV 1e-8f

// LDS tile: 256 block columns x 32 K-values, padded to 40 elems (80B stride:
// 16-lane fragment reads hit 16 distinct banks; 16B alignment preserved).
#define BCOLS 256
#define KSTEP 32
#define LPAD  40

// round-to-nearest-even f32 -> bf16 (pure integer path; always compiles)
static __device__ inline bf16_t f2bf(float f) {
    unsigned u = __builtin_bit_cast(unsigned, f);
    unsigned r = (u + 0x7FFFu + ((u >> 16) & 1u)) >> 16;
    unsigned short s = (unsigned short)r;
    return __builtin_bit_cast(bf16_t, s);
}

// A fragment (16x32 bf16): lane half h holds contiguous k = 16h..16h+15.
static __device__ inline v16bf load_a_bf16(const bf16_t* __restrict__ base, int half) {
    v8bf lo = *(const v8bf*)(base + half * 16);
    v8bf hi = *(const v8bf*)(base + half * 16 + 8);
    v16bf r;
#pragma unroll
    for (int i = 0; i < 8; ++i) { r[i] = lo[i]; r[8 + i] = hi[i]; }
    return r;
}

// B fragment from LDS (generic pointer into __shared__): two 16B ds reads.
static __device__ inline v16bf load_b_lds(const bf16_t* base) {
    v8bf lo = *(const v8bf*)(base);
    v8bf hi = *(const v8bf*)(base + 8);
    v16bf r;
#pragma unroll
    for (int i = 0; i < 8; ++i) { r[i] = lo[i]; r[8 + i] = hi[i]; }
    return r;
}

// CDNA5 async copy: LDS[lds_off] <- MEM[gptr], 16 bytes, tracked by ASYNCcnt.
static __device__ inline void async_ld_b128(unsigned lds_off, const bf16_t* gptr) {
    asm volatile("global_load_async_to_lds_b128 %0, %1, off"
                 :
                 : "v"(lds_off), "v"(gptr)
                 : "memory");
}
static __device__ inline void wait_async0() {
    asm volatile("s_wait_asynccnt 0x0" ::: "memory");
}

// Cooperative async fill of one LDS buffer: 256 threads x 4 x b128.
// src(col) = colbase + col*rowstride + k0, 32 bf16 per column.
static __device__ inline void fill_btile(const bf16_t* __restrict__ colbase,
                                         int k0, unsigned lds_buf_off, int tid) {
    const bf16_t* g = colbase + (size_t)tid * DD + k0;
    unsigned l = lds_buf_off + (unsigned)tid * (LPAD * 2);
#pragma unroll
    for (int c = 0; c < 4; ++c)
        async_ld_b128(l + c * 16, g + c * 8);
}

// ---------------------------------------------------------------------------
// 0) f32 -> bf16 streaming convert (vectorized: 8 elems/thread)
// ---------------------------------------------------------------------------
__global__ __launch_bounds__(256)
void k_cvt_bf16(const float* __restrict__ X, bf16_t* __restrict__ Y, int n) {
    int i = (blockIdx.x * 256 + threadIdx.x) * 8;
    if (i < n) {
        v4f a = *(const v4f*)(X + i);
        v4f b = *(const v4f*)(X + i + 4);
        v8bf o;
#pragma unroll
        for (int j = 0; j < 4; ++j) { o[j] = f2bf(a[j]); o[4 + j] = f2bf(b[j]); }
        *(v8bf*)(Y + i) = o;
    }
}

// ---------------------------------------------------------------------------
// 1) context pack: one f32 read ->
//      Crow[b][s][d] = bf16(context[b][s][d])      (A-operand of K-proj)
//      CT  [b][d][s] = bf16(context[b][s][d])      (B-operand of final GEMM)
// ---------------------------------------------------------------------------
__global__ __launch_bounds__(256)
void k_ctx_pack(const float* __restrict__ C, bf16_t* __restrict__ Crow,
                bf16_t* __restrict__ T) {
    __shared__ float tile[32][33];
    const int b = blockIdx.x, s0 = blockIdx.y * 32, d0 = blockIdx.z * 32;
    const int c = threadIdx.x & 31, r0 = threadIdx.x >> 5;
    const float* src = C + ((size_t)b * SL + s0) * DD + d0;
    bf16_t* rdst = Crow + ((size_t)b * SL + s0) * DD + d0;
#pragma unroll
    for (int i = 0; i < 32; i += 8) {
        float v = src[(size_t)(r0 + i) * DD + c];
        tile[r0 + i][c] = v;
        rdst[(size_t)(r0 + i) * DD + c] = f2bf(v);
    }
    __syncthreads();
    bf16_t* tdst = T + ((size_t)b * DD + d0) * SL + s0;
#pragma unroll
    for (int i = 0; i < 32; i += 8)
        tdst[(size_t)(r0 + i) * SL + c] = f2bf(tile[c][r0 + i]);
}

// ---------------------------------------------------------------------------
// 2) Projection GEMM: Y[m,n] = bf16( sum_d X[m,d]*W[n,d] + bias[n] )
//    Block = 8 waves as 4(M) x 2(N) -> 64 x 256 tile.
//    B tile async-staged into double-buffered LDS; 8 WMMAs/wave/K-step.
// ---------------------------------------------------------------------------
__global__ __launch_bounds__(256)
void k_proj_gemm(const bf16_t* __restrict__ X, const bf16_t* __restrict__ Wb,
                 const float* __restrict__ bias, bf16_t* __restrict__ Y) {
    __shared__ bf16_t sb[2][BCOLS][LPAD];
    const int tid  = threadIdx.x;
    const int lane = tid & 31;
    const int w    = tid >> 5;
    const int wm = w >> 1, wn = w & 1;
    const int m0 = blockIdx.x * 64 + wm * 16;
    const int n0 = blockIdx.y * 256 + wn * 128;
    const int nl = lane & 15, half = lane >> 4;

    const unsigned lds0 = (unsigned)(size_t)(&sb[0][0][0]);
    const unsigned lds1 = (unsigned)(size_t)(&sb[1][0][0]);
    const bf16_t* colbase = Wb + (size_t)(blockIdx.y * 256) * DD;

    v8f acc[8];
#pragma unroll
    for (int t = 0; t < 8; ++t) {
        float bv = bias[n0 + t * 16 + nl];
#pragma unroll
        for (int r = 0; r < 8; ++r) acc[t][r] = bv;
    }

    const bf16_t* arow = X + (size_t)(m0 + nl) * DD;

    fill_btile(colbase, 0, lds0, tid);
    wait_async0();
    __syncthreads();

    for (int k0 = 0; k0 < DD; k0 += KSTEP) {
        const int cur = (k0 >> 5) & 1;
        if (k0 + KSTEP < DD)
            fill_btile(colbase, k0 + KSTEP, cur ? lds0 : lds1, tid);

        __builtin_prefetch(arow + k0 + 256, 0, 1);
        v16bf a = load_a_bf16(arow + k0, half);
#pragma unroll
        for (int t = 0; t < 8; ++t) {
            v16bf bb = load_b_lds(&sb[cur][wn * 128 + t * 16 + nl][half * 16]);
            acc[t] = __builtin_amdgcn_wmma_f32_16x16x32_bf16(
                false, a, false, bb, (short)0, acc[t], false, false);
        }
        if (k0 + KSTEP < DD) wait_async0();
        __syncthreads();
    }

    const int mb = m0 + half * 8;
#pragma unroll
    for (int t = 0; t < 8; ++t) {
        int n = n0 + t * 16 + nl;
#pragma unroll
        for (int r = 0; r < 8; ++r)
            Y[(size_t)(mb + r) * DD + n] = f2bf(acc[t][r]);
    }
}

// ---------------------------------------------------------------------------
// 3) Scores: Sc[b][s][q] = LeakyReLU0.1( sum_e simK[b,s,e]*simQ[b,q,e] )
//    Same async-LDS staging of the Q (B-operand) tile.
// ---------------------------------------------------------------------------
__global__ __launch_bounds__(256)
void k_score_gemm(const bf16_t* __restrict__ Kb, const bf16_t* __restrict__ Qb,
                  float* __restrict__ Sc) {
    __shared__ bf16_t sb[2][BCOLS][LPAD];
    const int b = blockIdx.x;
    const int tid  = threadIdx.x;
    const int lane = tid & 31;
    const int w    = tid >> 5;
    const int wm = w >> 1, wn = w & 1;
    const int s0 = blockIdx.y * 64 + wm * 16;
    const int q0 = blockIdx.z * 256 + wn * 128;
    const int nl = lane & 15, half = lane >> 4;

    const unsigned lds0 = (unsigned)(size_t)(&sb[0][0][0]);
    const unsigned lds1 = (unsigned)(size_t)(&sb[1][0][0]);
    const bf16_t* colbase = Qb + ((size_t)b * QL + blockIdx.z * 256) * DD;

    v8f acc[8] = {v8f{}, v8f{}, v8f{}, v8f{}, v8f{}, v8f{}, v8f{}, v8f{}};

    const bf16_t* arow = Kb + ((size_t)b * SL + s0 + nl) * DD;

    fill_btile(colbase, 0, lds0, tid);
    wait_async0();
    __syncthreads();

    for (int k0 = 0; k0 < DD; k0 += KSTEP) {
        const int cur = (k0 >> 5) & 1;
        if (k0 + KSTEP < DD)
            fill_btile(colbase, k0 + KSTEP, cur ? lds0 : lds1, tid);

        v16bf a = load_a_bf16(arow + k0, half);
#pragma unroll
        for (int t = 0; t < 8; ++t) {
            v16bf bb = load_b_lds(&sb[cur][wn * 128 + t * 16 + nl][half * 16]);
            acc[t] = __builtin_amdgcn_wmma_f32_16x16x32_bf16(
                false, a, false, bb, (short)0, acc[t], false, false);
        }
        if (k0 + KSTEP < DD) wait_async0();
        __syncthreads();
    }

    float* obase = Sc + (size_t)b * SL * QL;
    const int mb = s0 + half * 8;
#pragma unroll
    for (int t = 0; t < 8; ++t) {
        int q = q0 + t * 16 + nl;
#pragma unroll
        for (int r = 0; r < 8; ++r) {
            float v = acc[t][r];
            v = (v >= 0.0f) ? v : 0.1f * v;   // LeakyReLU(0.1)
            obase[(size_t)(mb + r) * QL + q] = v;
        }
    }
}

// ---------------------------------------------------------------------------
// 4) Per-(b,s) inverse L2 norm over q:  inv[row] = 1/(sqrt(sum_q Sc^2)+eps)
// ---------------------------------------------------------------------------
__global__ __launch_bounds__(256)
void k_row_sumsq(const float* __restrict__ Sc, float* __restrict__ inv) {
    const size_t row = blockIdx.x;              // b*SL + s
    const float* p = Sc + row * QL;
    const int t = threadIdx.x;
    float v0 = p[t], v1 = p[t + 256];
    float s = v0 * v0 + v1 * v1;
#pragma unroll
    for (int m = 16; m >= 1; m >>= 1) s += __shfl_xor(s, m, 32);
    __shared__ float red[8];
    if ((t & 31) == 0) red[t >> 5] = s;
    __syncthreads();
    if (t == 0) {
        float tot = 0.0f;
#pragma unroll
        for (int i = 0; i < 8; ++i) tot += red[i];
        inv[row] = 1.0f / (sqrtf(tot) + EPSV);
    }
}

// ---------------------------------------------------------------------------
// 5) Transpose + scale: attT[b][q][s] = Sc[b][s][q] * inv[b,s] * smooth
// ---------------------------------------------------------------------------
__global__ __launch_bounds__(256)
void k_score_transpose(const float* __restrict__ Sc, const float* __restrict__ inv,
                       const float* __restrict__ smooth_p, float* __restrict__ T) {
    __shared__ float tile[32][33];
    const int b = blockIdx.x, s0 = blockIdx.y * 32, q0 = blockIdx.z * 32;
    const float smooth = smooth_p[0];
    const int c = threadIdx.x & 31, r0 = threadIdx.x >> 5;
    const float* src = Sc + ((size_t)b * SL + s0) * QL + q0;
#pragma unroll
    for (int i = 0; i < 32; i += 8) {
        int s = r0 + i;
        tile[s][c] = src[(size_t)s * QL + c] * inv[(size_t)b * SL + s0 + s] * smooth;
    }
    __syncthreads();
    float* dst = T + ((size_t)b * QL + q0) * SL + s0;
#pragma unroll
    for (int i = 0; i < 32; i += 8)
        dst[(size_t)(r0 + i) * SL + c] = tile[c][r0 + i];
}

// ---------------------------------------------------------------------------
// 6) Softmax over s (row length SL=1024), output bf16 attn[b][q][s]
// ---------------------------------------------------------------------------
__global__ __launch_bounds__(256)
void k_softmax_rows(const float* __restrict__ T, bf16_t* __restrict__ A) {
    const size_t row = blockIdx.x;              // b*QL + q
    const float* p = T + row * SL;
    const int t = threadIdx.x;
    float v[4];
    float mx = -3.4e38f;
#pragma unroll
    for (int i = 0; i < 4; ++i) { v[i] = p[t + i * 256]; mx = fmaxf(mx, v[i]); }
#pragma unroll
    for (int m = 16; m >= 1; m >>= 1) mx = fmaxf(mx, __shfl_xor(mx, m, 32));
    __shared__ float red[8];
    if ((t & 31) == 0) red[t >> 5] = mx;
    __syncthreads();
    float gmx = red[0];
#pragma unroll
    for (int i = 1; i < 8; ++i) gmx = fmaxf(gmx, red[i]);
    __syncthreads();
    float s = 0.0f;
#pragma unroll
    for (int i = 0; i < 4; ++i) { v[i] = __expf(v[i] - gmx); s += v[i]; }
#pragma unroll
    for (int m = 16; m >= 1; m >>= 1) s += __shfl_xor(s, m, 32);
    if ((t & 31) == 0) red[t >> 5] = s;
    __syncthreads();
    float tot = 0.0f;
#pragma unroll
    for (int i = 0; i < 8; ++i) tot += red[i];
    const float invs = 1.0f / tot;
    bf16_t* o = A + row * SL;
#pragma unroll
    for (int i = 0; i < 4; ++i) o[t + i * 256] = f2bf(v[i] * invs);
}

// ---------------------------------------------------------------------------
// 7) Final: O[b][q][d] = l2norm_d( sum_s attn[b,q,s] * context[b,s,d] )
//    One block per (b, 16-q tile); 8 waves each own 128 of the 1024 d's,
//    whole output row lives in the block -> fused L2 norm epilogue.
// ---------------------------------------------------------------------------
__global__ __launch_bounds__(256)
void k_out_gemm(const bf16_t* __restrict__ A, const bf16_t* __restrict__ CT,
                float* __restrict__ O) {
    const int bq = blockIdx.x;                  // b*(QL/16) + qtile
    const int b  = bq >> 5;                     // QL/16 == 32
    const int q0 = (bq & 31) * 16;
    const int lane = threadIdx.x & 31;
    const int w    = threadIdx.x >> 5;
    const int nl = lane & 15, half = lane >> 4;
    const int n_base = w * 128;

    v8f acc[8] = {v8f{}, v8f{}, v8f{}, v8f{}, v8f{}, v8f{}, v8f{}, v8f{}};

    const bf16_t* arow = A + ((size_t)b * QL + q0 + nl) * SL;
    const bf16_t* brow[8];
#pragma unroll
    for (int t = 0; t < 8; ++t)
        brow[t] = CT + ((size_t)b * DD + n_base + t * 16 + nl) * SL + half * 16;

    for (int k0 = 0; k0 < SL; k0 += 32) {
        v16bf af = load_a_bf16(arow + k0, half);
        v16bf bb[8];
#pragma unroll
        for (int t = 0; t < 8; ++t)
            bb[t] = *(const v16bf*)(brow[t] + k0);
#pragma unroll
        for (int t = 0; t < 8; ++t)
            acc[t] = __builtin_amdgcn_wmma_f32_16x16x32_bf16(
                false, af, false, bb[t], (short)0, acc[t], false, false);
    }

    // fused L2 norm over d: rows (q0 + half*8 + r), r = 0..7
    __shared__ float ssq[8][16];
    const int mb = half * 8;
    float part[8];
#pragma unroll
    for (int r = 0; r < 8; ++r) {
        float s = 0.0f;
#pragma unroll
        for (int t = 0; t < 8; ++t) s += acc[t][r] * acc[t][r];
#pragma unroll
        for (int m = 1; m < 16; m <<= 1) s += __shfl_xor(s, m, 32);  // within 16-lane group
        part[r] = s;
    }
    if (nl == 0) {
#pragma unroll
        for (int r = 0; r < 8; ++r) ssq[w][mb + r] = part[r];
    }
    __syncthreads();
    float scale[8];
#pragma unroll
    for (int r = 0; r < 8; ++r) {
        float tot = 0.0f;
#pragma unroll
        for (int ww = 0; ww < 8; ++ww) tot += ssq[ww][mb + r];
        scale[r] = 1.0f / (sqrtf(tot) + EPSV);
    }
    float* orow = O + ((size_t)b * QL + q0) * DD;
#pragma unroll
    for (int t = 0; t < 8; ++t) {
        int n = n_base + t * 16 + nl;
#pragma unroll
        for (int r = 0; r < 8; ++r)
            orow[(size_t)(mb + r) * DD + n] = acc[t][r] * scale[r];
    }
}

// ---------------------------------------------------------------------------
// launcher
// ---------------------------------------------------------------------------
extern "C" void kernel_launch(void* const* d_in, const int* in_sizes, int n_in,
                              void* d_out, int out_size, void* d_ws, size_t ws_size,
                              hipStream_t stream) {
    const float* query   = (const float*)d_in[0];   // (B, QL, D)
    const float* context = (const float*)d_in[1];   // (B, SL, D)
    const float* Wq      = (const float*)d_in[2];   // (D, D)
    const float* bq      = (const float*)d_in[3];   // (D,)
    const float* Wk      = (const float*)d_in[4];   // (D, D)
    const float* bk      = (const float*)d_in[5];   // (D,)
    const float* smooth  = (const float*)d_in[6];   // scalar
    float* out = (float*)d_out;                     // (B, QL, D)

    char* ws = (char*)d_ws;
    size_t off = 0;
    auto carve = [&](size_t bytes) -> char* {
        char* p = ws + off;
        off += (bytes + 255) & ~(size_t)255;
        return p;
    };
    bf16_t* Wqb  = (bf16_t*)carve((size_t)DD * DD * 2);          //   2 MB
    bf16_t* Wkb  = (bf16_t*)carve((size_t)DD * DD * 2);          //   2 MB
    bf16_t* qb   = (bf16_t*)carve((size_t)Bx * QL * DD * 2);     // 134 MB \ scores (268MB) aliases
    bf16_t* cb   = (bf16_t*)carve((size_t)Bx * SL * DD * 2);     // 268 MB / this region later
    bf16_t* simq = (bf16_t*)carve((size_t)Bx * QL * DD * 2);     // 134 MB (later: attn bf16)
    bf16_t* simk = (bf16_t*)carve((size_t)Bx * SL * DD * 2);     // 268 MB (later: attT f32)
    bf16_t* ctxT = (bf16_t*)carve((size_t)Bx * DD * SL * 2);     // 268 MB
    float*  inv  = (float*) carve((size_t)Bx * SL * 4);          // 0.5 MB
    float*  score = (float*)(void*)qb;    // alias qb+cb (402MB region, need 268MB; dead after proj)
    float*  attT  = (float*)(void*)simk;  // alias (simk dead after scores)
    bf16_t* attnb = simq;                 // alias (simq dead after scores)

    // 0) weights + query -> bf16 (streaming)
    k_cvt_bf16<<<dim3((DD * DD) / 2048), dim3(256), 0, stream>>>(Wq, Wqb, DD * DD);
    k_cvt_bf16<<<dim3((DD * DD) / 2048), dim3(256), 0, stream>>>(Wk, Wkb, DD * DD);
    k_cvt_bf16<<<dim3((Bx * QL * DD) / 2048), dim3(256), 0, stream>>>(query, qb, Bx * QL * DD);

    // 1) context -> bf16 row-major + (b,d,s) transposed, single f32 read
    k_ctx_pack<<<dim3(Bx, SL / 32, DD / 32), dim3(256), 0, stream>>>(context, cb, ctxT);

    // 2) projections (bf16 WMMA, f32 accumulate, bias fused, async-LDS B tile)
    k_proj_gemm<<<dim3((Bx * QL) / 64, DD / 256), dim3(256), 0, stream>>>(qb, Wqb, bq, simq);
    k_proj_gemm<<<dim3((Bx * SL) / 64, DD / 256), dim3(256), 0, stream>>>(cb, Wkb, bk, simk);

    // 3) scores + LeakyReLU (async-LDS B tile)
    k_score_gemm<<<dim3(Bx, SL / 64, QL / 256), dim3(256), 0, stream>>>(simk, simq, score);

    // 4) per-(b,s) L2 norm over q
    k_row_sumsq<<<dim3(Bx * SL), dim3(256), 0, stream>>>(score, inv);

    // 5) transpose + scale by inv-norm and smooth
    k_score_transpose<<<dim3(Bx, SL / 32, QL / 32), dim3(256), 0, stream>>>(score, inv, smooth, attT);

    // 6) softmax over s -> bf16 attention
    k_softmax_rows<<<dim3(Bx * QL), dim3(256), 0, stream>>>(attT, attnb);

    // 7) weighted sum over context + fused final L2 norm
    k_out_gemm<<<dim3(Bx * (QL / 16)), dim3(256), 0, stream>>>(attnb, ctxT, out);

    (void)in_sizes; (void)n_in; (void)out_size; (void)ws_size;
}
